// Attention_56392920596865
// MI455X (gfx1250) — matrix operand order repaired and checked by reading, not compiled, not run on hardware
//
#include <hip/hip_runtime.h>
#include <hip/hip_bf16.h>

// Problem constants (from the reference)
#define BB  4
#define SS  2048
#define DD  1024
#define HH  16
#define HDD 64

typedef __attribute__((ext_vector_type(16))) __bf16         v16bf;
typedef __attribute__((ext_vector_type(8)))  float          v8f;
typedef __attribute__((ext_vector_type(8)))  unsigned short v8u;
typedef __attribute__((ext_vector_type(4)))  float          v4f;
typedef __attribute__((ext_vector_type(4)))  unsigned short v4u;
typedef __attribute__((ext_vector_type(4)))  unsigned int   v4u32;
typedef __attribute__((ext_vector_type(8)))  int            v8i32;
typedef __attribute__((ext_vector_type(4)))  int            v4i32;

// ---- fp32 <-> bf16 (round-to-nearest-even)
__device__ __forceinline__ unsigned short f2bfu(float f) {
  unsigned int u = __builtin_bit_cast(unsigned int, f);
  u += 0x7fffu + ((u >> 16) & 1u);
  return (unsigned short)(u >> 16);
}
__device__ __forceinline__ float bfu2f(unsigned short h) {
  unsigned int u = ((unsigned int)h) << 16;
  return __builtin_bit_cast(float, u);
}

union FragAB {
  v16bf v;
  v8u   h[2];
  unsigned short u[16];
};

// =====================================================================
// bf16 WMMA GEMM: C(MxN) = A(MxK) * B(NxK)^T  (both operands K-contiguous)
// Workgroup = 4 waves (128 thr), wave tile 64x64, WG tile 128x128, K step 32.
// MODE: 0 = projection: scatter bf16 via head_shuffle; if WVT, write ONLY the
//           transposed V^T[b][d][s] copy (for the PV GEMM's B operand)
//       1 = attention scores, *1/32, bf16 out, batched (z)
//       2 = PV, scatter bf16 via inverse head shuffle, batched (z)
//       3 = plain f32 store (final output)
// =====================================================================
template <int MODE, bool WVT>
__global__ __launch_bounds__(128)
void gemm_wmma(const unsigned short* __restrict__ Ab,
               const unsigned short* __restrict__ Bb,
               void* __restrict__ Cb,
               unsigned short* __restrict__ VTb,
               int lda, int ldb, int Kdim,
               long long aStride, long long bStride) {
  const int lane  = threadIdx.x & 31;
  const int wave  = threadIdx.x >> 5;
  const int batch = blockIdx.z;
  const int rowBase = blockIdx.y * 128 + (wave >> 1) * 64;
  const int colBase = blockIdx.x * 128 + (wave & 1) * 64;

  const unsigned short* Am = Ab + (size_t)batch * (size_t)aStride;
  const unsigned short* Bm = Bb + (size_t)batch * (size_t)bStride;

  const int l  = lane & 15;   // row/col within 16-strip
  const int hi = lane >> 4;   // lane-half selector (ISA 7.12.2 layouts)

  const v8f vzero = {};
  v8f acc[4][4];
#pragma unroll
  for (int i = 0; i < 4; ++i)
#pragma unroll
    for (int j = 0; j < 4; ++j) acc[i][j] = vzero;

  for (int kb = 0; kb < Kdim; kb += 32) {
    FragAB a[4], b[4];
    // A fragments (16-bit A 16x32 layout):
    //   lanes 0-15:  K = kb+{0..7, 16..23}; lanes 16-31: K = kb+{8..15, 24..31}
#pragma unroll
    for (int i = 0; i < 4; ++i) {
      const unsigned short* p =
          Am + (size_t)(rowBase + i * 16 + l) * lda + kb + hi * 8;
      a[i].h[0] = *(const v8u*)(p);
      a[i].h[1] = *(const v8u*)(p + 16);
    }
    // B fragments: lane holds one column, same K order as A
    //   lanes 0-15: K = kb+{0..7, 16..23}; lanes 16-31: K = kb+{8..15, 24..31}
#pragma unroll
    for (int j = 0; j < 4; ++j) {
      const unsigned short* p =
          Bm + (size_t)(colBase + j * 16 + l) * ldb + kb + hi * 8;
      b[j].h[0] = *(const v8u*)(p);
      b[j].h[1] = *(const v8u*)(p + 16);
    }
    if (kb + 32 < Kdim) {
      // locality=3 -> WGP-scope prefetch (pulls into all cache levels)
      __builtin_prefetch(Bm + (size_t)(colBase + l) * ldb + kb + 32, 0, 3);
      __builtin_prefetch(Am + (size_t)(rowBase + l) * lda + kb + 32, 0, 3);
    }
#pragma unroll
    for (int i = 0; i < 4; ++i)
#pragma unroll
      for (int j = 0; j < 4; ++j)
        acc[i][j] = __builtin_amdgcn_wmma_f32_16x16x32_bf16(
            false, a[i].v, false, b[j].v, (short)0, acc[i][j], false, false);
  }

  // Epilogue. C/D layout: VGPR r -> M = r (+8 for lanes 16-31), N = lane%16
#pragma unroll
  for (int i = 0; i < 4; ++i) {
#pragma unroll
    for (int j = 0; j < 4; ++j) {
#pragma unroll
      for (int r = 0; r < 8; ++r) {
        const int m = rowBase + i * 16 + hi * 8 + r;
        const int n = colBase + j * 16 + l;
        const float val = acc[i][j][r];
        if constexpr (MODE == 0) {
          // head_shuffle: dest flat = s*H*B*HD + h*B*HD + b*HD + hd
          const int bi = m >> 11;          // m / S
          const int s  = m & (SS - 1);
          const int h  = n >> 6;
          const int hd = n & 63;
          const size_t dst = (size_t)s * 4096 + (size_t)h * 256 + bi * 64 + hd;
          const unsigned short bv = f2bfu(val);
          if constexpr (WVT) {
            // V path: only the transposed copy V^T[b2][d2][s2] is consumed
            const int b2 = (int)(dst >> 21);               // / (S*D)
            const size_t rem = dst & ((1ull << 21) - 1);
            const int s2 = (int)(rem >> 10);
            const int d2 = (int)(rem & (DD - 1));
            VTb[((size_t)b2 * DD + d2) * SS + s2] = bv;
          } else {
            ((unsigned short*)Cb)[dst] = bv;
          }
        } else if constexpr (MODE == 1) {
          ((unsigned short*)Cb)[(size_t)batch * SS * SS + (size_t)m * SS + n] =
              f2bfu(val * 0.03125f);  // 1/sqrt(1024)
        } else if constexpr (MODE == 2) {
          // inverse shuffle: flat over whole (B,S,D); view (H,B,S,HD)->(B,S,H,HD)
          const size_t flat = ((size_t)batch * SS + m) * DD + n;
          const int hd  = (int)(flat & 63);
          const size_t t = flat >> 6;
          const int s   = (int)(t & (SS - 1));
          const int bb2 = (int)((t >> 11) & 3);
          const int h   = (int)(t >> 13);
          ((unsigned short*)Cb)[((size_t)bb2 * SS + s) * DD + h * 64 + hd] =
              f2bfu(val);
        } else {
          ((float*)Cb)[(size_t)m * DD + n] = val;
        }
      }
    }
  }
}

// ---- fp32 -> bf16 bulk convert, 4-wide
__global__ __launch_bounds__(256)
void cvt4(const float* __restrict__ in, unsigned short* __restrict__ out, int n4) {
  const int i = blockIdx.x * 256 + threadIdx.x;
  if (i < n4) {
    v4f x = *(const v4f*)(in + (size_t)4 * i);
    v4u r;
    r[0] = f2bfu(x[0]); r[1] = f2bfu(x[1]); r[2] = f2bfu(x[2]); r[3] = f2bfu(x[3]);
    *(v4u*)(out + (size_t)4 * i) = r;
  }
}

// ---- row softmax over 2048 bf16 scores; row is DMA'd to LDS by the Tensor
//      Data Mover (TENSORcnt-tracked), f32 math, bf16 written back in place.
__global__ __launch_bounds__(256)
void softmax_rows(unsigned short* __restrict__ Scb) {
  const int row = blockIdx.x;                 // 0 .. B*S-1
  unsigned short* rp = Scb + (size_t)row * SS;
  const int t = threadIdx.x;

  __shared__ unsigned short rowbuf[SS];       // 4 KB tile target
  __shared__ float red[256];

  if (t < 32) {  // wave 0 issues the TDM load (one descriptor per workgroup)
    const unsigned long long ga = (unsigned long long)(uintptr_t)rp;
    const unsigned lds_off = (unsigned)(uintptr_t)(&rowbuf[0]);
    v4u32 g0;
    g0[0] = 1u;                                    // count=1, user mode
    g0[1] = lds_off;                               // D#.lds_addr
    g0[2] = (unsigned)(ga & 0xffffffffu);          // global_addr[31:0]
    g0[3] = (unsigned)((ga >> 32) & 0x01ffffffu)   // global_addr[56:32]
            | (2u << 30);                          // type=2 ("image")
    v8i32 g1;
    g1[0] = (int)(1u << 16);          // wg_mask=0, data_size=1 (2 bytes)
    g1[1] = (int)(2048u << 16);       // tensor_dim0[15:0]=2048 @ bits63:48
    g1[2] = (int)(0u | (1u << 16));   // tensor_dim0[31:16]=0; tensor_dim1[15:0]=1
    g1[3] = (int)(2048u << 16);       // tensor_dim1[31:16]=0; tile_dim0=2048
    g1[4] = 1;                        // tile_dim1=1, tile_dim2=0
    g1[5] = 2048;                     // tensor_dim0_stride[31:0]
    g1[6] = 0;                        // dim0_stride[47:32]=0, dim1_stride[15:0]=0
    g1[7] = 0;
    const v4i32 gz4 = {0, 0, 0, 0};   // groups 2/3 unused (2D tensor)
    const v8i32 gz8 = {0, 0, 0, 0, 0, 0, 0, 0};
    // 6-arg toolchain form: (g0, g1, g2, g3, extra, cpol)
    __builtin_amdgcn_tensor_load_to_lds(g0, g1, gz4, gz4, gz8, 0);
    __builtin_amdgcn_s_wait_tensorcnt(0);
  }
  __syncthreads();

  float v[8];
  float mx = -3.0e38f;
#pragma unroll
  for (int i = 0; i < 8; ++i) { v[i] = bfu2f(rowbuf[t + 256 * i]); mx = fmaxf(mx, v[i]); }
  red[t] = mx; __syncthreads();
  for (int s = 128; s > 0; s >>= 1) { if (t < s) red[t] = fmaxf(red[t], red[t + s]); __syncthreads(); }
  mx = red[0]; __syncthreads();
  float sum = 0.f;
#pragma unroll
  for (int i = 0; i < 8; ++i) { v[i] = __expf(v[i] - mx); sum += v[i]; }
  red[t] = sum; __syncthreads();
  for (int s = 128; s > 0; s >>= 1) { if (t < s) red[t] += red[t + s]; __syncthreads(); }
  const float inv = 1.f / red[0];
#pragma unroll
  for (int i = 0; i < 8; ++i) rp[t + 256 * i] = f2bfu(v[i] * inv);
}

extern "C" void kernel_launch(void* const* d_in, const int* in_sizes, int n_in,
                              void* d_out, int out_size, void* d_ws, size_t ws_size,
                              hipStream_t stream) {
  (void)in_sizes; (void)n_in; (void)out_size; (void)ws_size;
  const float* k_in = (const float*)d_in[0];
  const float* q_in = (const float*)d_in[1];
  const float* v_in = (const float*)d_in[2];
  const float* Wk   = (const float*)d_in[3];
  const float* Wq   = (const float*)d_in[4];
  const float* Wv   = (const float*)d_in[5];
  const float* Wo   = (const float*)d_in[6];

  const size_t DDE = (size_t)DD * DD;        // 1M  elements
  const size_t BSD = (size_t)BB * SS * DD;   // 8M  elements
  const size_t BSS = (size_t)BB * SS * SS;   // 16M elements

  // workspace carve (all bf16): 76M elems = 152 MB
  unsigned short* p = (unsigned short*)d_ws;
  unsigned short* Wk_b = p; p += DDE;
  unsigned short* Wq_b = p; p += DDE;
  unsigned short* Wv_b = p; p += DDE;
  unsigned short* Wo_b = p; p += DDE;
  unsigned short* Kb  = p; p += BSD;   // bf16 activations
  unsigned short* Qb  = p; p += BSD;
  unsigned short* Vb  = p; p += BSD;
  unsigned short* Kp  = p; p += BSD;   // shuffled K projection
  unsigned short* Qp  = p; p += BSD;   // shuffled Q projection
  unsigned short* VT  = p; p += BSD;   // V^T per batch [b][d][s]
  unsigned short* Ob  = p; p += BSD;   // attention out, un-shuffled ("mh") layout
  unsigned short* Scb = p; p += BSS;   // scores -> softmax weights (in place)

  const dim3 c256(256);
  const int w4 = (int)(DDE / 4);
  const int a4 = (int)(BSD / 4);
  cvt4<<<dim3((unsigned)(w4 / 256)), c256, 0, stream>>>(Wk, Wk_b, w4);
  cvt4<<<dim3((unsigned)(w4 / 256)), c256, 0, stream>>>(Wq, Wq_b, w4);
  cvt4<<<dim3((unsigned)(w4 / 256)), c256, 0, stream>>>(Wv, Wv_b, w4);
  cvt4<<<dim3((unsigned)(w4 / 256)), c256, 0, stream>>>(Wo, Wo_b, w4);
  cvt4<<<dim3((unsigned)(a4 / 256)), c256, 0, stream>>>(k_in, Kb, a4);
  cvt4<<<dim3((unsigned)(a4 / 256)), c256, 0, stream>>>(q_in, Qb, a4);
  cvt4<<<dim3((unsigned)(a4 / 256)), c256, 0, stream>>>(v_in, Vb, a4);

  const dim3 blk(128);
  // projections: M=B*S=8192, N=D=1024, K=D=1024
  gemm_wmma<0, false><<<dim3(8, 64, 1), blk, 0, stream>>>(
      Kb, Wk_b, Kp, nullptr, DD, DD, DD, 0, 0);
  gemm_wmma<0, false><<<dim3(8, 64, 1), blk, 0, stream>>>(
      Qb, Wq_b, Qp, nullptr, DD, DD, DD, 0, 0);
  gemm_wmma<0, true ><<<dim3(8, 64, 1), blk, 0, stream>>>(
      Vb, Wv_b, nullptr, VT, DD, DD, DD, 0, 0);
  // scores: per batch, M=N=S=2048, K=D=1024
  gemm_wmma<1, false><<<dim3(16, 16, 4), blk, 0, stream>>>(
      Qp, Kp, Scb, nullptr, DD, DD, DD,
      (long long)SS * DD, (long long)SS * DD);
  softmax_rows<<<dim3((unsigned)(BB * SS)), c256, 0, stream>>>(Scb);
  // PV: per batch, M=S=2048, N=D=1024, K=S=2048 (B operand = V^T, K-contig)
  gemm_wmma<2, false><<<dim3(8, 16, 4), blk, 0, stream>>>(
      Scb, VT, Ob, nullptr, SS, SS, SS,
      (long long)SS * SS, (long long)DD * SS);
  // output projection: M=8192, N=1024, K=1024, fp32 out
  gemm_wmma<3, false><<<dim3(8, 64, 1), blk, 0, stream>>>(
      Ob, Wo_b, d_out, nullptr, DD, DD, DD, 0, 0);
}
